// Attn_81114752352943
// MI455X (gfx1250) — matrix operand order repaired and checked by reading, not compiled, not run on hardware
//
#include <hip/hip_runtime.h>

// ---------------------------------------------------------------------------
// Types for CDNA5 WMMA (wave32): v_wmma_f32_16x16x32_bf16
// ---------------------------------------------------------------------------
typedef __attribute__((ext_vector_type(16))) __bf16 bf16x16;
typedef __attribute__((ext_vector_type(8)))  __bf16 bf16x8;
typedef __attribute__((ext_vector_type(8)))  float  f32x8;

#define T_SEQ 2048
#define DM    2048
#define NH    16
#define NKV   4
#define DH    128

#if __has_builtin(__builtin_amdgcn_tensor_load_to_lds) && \
    __has_builtin(__builtin_amdgcn_s_wait_tensorcnt)
#define HAVE_TDM 1
#endif

#if defined(HAVE_TDM)
typedef unsigned int u32x4 __attribute__((ext_vector_type(4)));
typedef int          i32x4 __attribute__((ext_vector_type(4)));
typedef int          i32x8 __attribute__((ext_vector_type(8)));
#endif

__device__ __forceinline__ f32x8 wmma_bf16(bf16x16 a, bf16x16 b, f32x8 c) {
    // 8 args: (neg_a, A, neg_b, B, c_mod, C, reuse_a, reuse_b)
    return __builtin_amdgcn_wmma_f32_16x16x32_bf16(
        false, a, false, b, (short)0, c, false, false);
}

// A-fragment (16x32 bf16, M in lanes, K in VGPRs). half = lane>>4.
//   element i<8  -> K = 8*half + i ; element i>=8 -> K = 16 + 8*half + (i-8)
__device__ __forceinline__ bf16x16 load_a_frag(const __bf16* rowp, int half) {
    const bf16x8 lo = *(const bf16x8*)(rowp + 8 * half);
    const bf16x8 hi = *(const bf16x8*)(rowp + 16 + 8 * half);
    bf16x16 a;
#pragma unroll
    for (int i = 0; i < 8; ++i) { a[i] = lo[i]; a[i + 8] = hi[i]; }
    return a;
}

// B-fragment (32x16 bf16, N in lanes, K in VGPRs):
//   element i<8  -> K = 8*half + i ; element i>=8 -> K = 16 + 8*half + (i-8)
__device__ __forceinline__ bf16x16 load_b_frag(const __bf16* colp, int half) {
    const bf16x8 lo = *(const bf16x8*)(colp + 8 * half);
    const bf16x8 hi = *(const bf16x8*)(colp + 16 + 8 * half);
    bf16x16 b;
#pragma unroll
    for (int i = 0; i < 8; ++i) { b[i] = lo[i]; b[i + 8] = hi[i]; }
    return b;
}

#if defined(HAVE_TDM)
// Issue a TDM load of a 64-row x 32-col bf16 tile of BT (row stride K elems)
// into LDS at byte offset lds_off. D# layout per CDNA5 ISA 8.3/8.4.
// This toolchain exposes the 6-arg builtin:
//   (u32x4 g0, i32x8 g1, i32x4 g2, i32x4 g3, i32x8 g4, i32 cpol)
__device__ __forceinline__ void tdm_load_b_tile(const __bf16* BT, int n0, int k0,
                                                int K, int N, unsigned lds_off) {
    const unsigned long long gaddr =
        (unsigned long long)(uintptr_t)(BT + (size_t)n0 * K + k0);
    u32x4 g0;
    g0[0] = 1u;                                            // count=1 (valid), user mode
    g0[1] = lds_off;                                       // lds_addr (bytes)
    g0[2] = (unsigned)(gaddr & 0xFFFFFFFFull);             // global_addr[31:0]
    g0[3] = (unsigned)((gaddr >> 32) & 0x1FFFFFFull)       // global_addr[56:32]
            | 0x80000000u;                                 // type=2 ("image")
    i32x8 g1;
    g1[0] = 1 << 16;                       // workgroup_mask=0, data_size=1 (2 bytes)
    g1[1] = (int)(((unsigned)K & 0xFFFFu) << 16);          // tensor_dim0[15:0]
    g1[2] = (int)(((unsigned)N & 0xFFFFu) << 16);          // dim0[31:16]=0, tensor_dim1[15:0]
    g1[3] = (int)(32u << 16);              // tensor_dim1[31:16]=0, tile_dim0=32
    g1[4] = 64;                            // tile_dim1=64, tile_dim2=0
    g1[5] = K;                             // tensor_dim0_stride[31:0]
    g1[6] = 0;                             // stride0[47:32]=0, dim1_stride[15:0]=0
    g1[7] = 0;
    i32x4 g2; g2[0] = 0; g2[1] = 0; g2[2] = 0; g2[3] = 0;  // unused (2D tensor)
    i32x4 g3; g3[0] = 0; g3[1] = 0; g3[2] = 0; g3[3] = 0;
    i32x8 g4; g4[0] = 0; g4[1] = 0; g4[2] = 0; g4[3] = 0;
              g4[4] = 0; g4[5] = 0; g4[6] = 0; g4[7] = 0;  // unused trailing group
    __builtin_amdgcn_tensor_load_to_lds(g0, g1, g2, g3, g4, 0);
}
#endif

// ---------------------------------------------------------------------------
// Elementwise converts
// ---------------------------------------------------------------------------
__global__ __launch_bounds__(256)
void cvt_bf16_kernel(const float* __restrict__ src, __bf16* __restrict__ dst, int n) {
    int i = blockIdx.x * 256 + threadIdx.x;
    if (i < n) dst[i] = (__bf16)src[i];
}

// W (K,N) row-major fp32  ->  WT (N,K) row-major bf16
__global__ __launch_bounds__(256)
void transpose_bf16_kernel(const float* __restrict__ W, __bf16* __restrict__ WT,
                           int K, int N) {
    int i = blockIdx.x * 256 + threadIdx.x;
    if (i < K * N) {
        int n = i % N;
        int k = i / N;
        WT[(size_t)n * K + k] = (__bf16)W[i];
    }
}

// ---------------------------------------------------------------------------
// bf16 WMMA GEMM: C(M,N) = A(M,K) @ B(K,N), B supplied transposed (N,K).
// Block = 8 waves; wave computes 32(M) x 64(N); block tile 256(M) x 64(N).
// B tile (64x32, 4KB) staged to LDS via TDM (double-buffered); A direct.
// grid = (N/64, M/256)
// ---------------------------------------------------------------------------
__global__ __launch_bounds__(256)
void gemm_bf16_kernel(const __bf16* __restrict__ A, const __bf16* __restrict__ BT,
                      float* __restrict__ C, int M, int N, int K) {
    __shared__ __bf16 sB[2][64 * 32];
    const int tid  = threadIdx.x;
    const int lane = tid & 31;
    const int wave = tid >> 5;
    const int half = lane >> 4;
    const int l16  = lane & 15;
    const int m0 = blockIdx.y * 256 + wave * 32;
    const int n0 = blockIdx.x * 64;

    f32x8 c[2][4];
#pragma unroll
    for (int ms = 0; ms < 2; ++ms)
#pragma unroll
        for (int t = 0; t < 4; ++t)
#pragma unroll
            for (int r = 0; r < 8; ++r) c[ms][t][r] = 0.0f;

    const __bf16* arow0 = A + (size_t)(m0 + l16) * K;
    const __bf16* arow1 = arow0 + (size_t)16 * K;

    // --- stage first B tile ---
#if defined(HAVE_TDM)
    if (wave == 0)
        tdm_load_b_tile(BT, n0, 0, K, N, (unsigned)(uintptr_t)&sB[0][0]);
#else
    {
        const int scb = tid >> 2, seb = (tid & 3) * 8;
        *(bf16x8*)&sB[0][scb * 32 + seb] =
            *(const bf16x8*)(BT + (size_t)(n0 + scb) * K + seb);
    }
#endif

    const int nsteps = K >> 5;
    for (int step = 0; step < nsteps; ++step) {
        const int k0 = step << 5;
        const int cur = step & 1;
        // --- stage next tile into the other buffer ---
        if (step + 1 < nsteps) {
#if defined(HAVE_TDM)
            if (wave == 0)
                tdm_load_b_tile(BT, n0, k0 + 32, K, N,
                                (unsigned)(uintptr_t)&sB[cur ^ 1][0]);
#else
            const int scb = tid >> 2, seb = (tid & 3) * 8;
            *(bf16x8*)&sB[cur ^ 1][scb * 32 + seb] =
                *(const bf16x8*)(BT + (size_t)(n0 + scb) * K + k0 + 32 + seb);
#endif
        }
#if defined(HAVE_TDM)
        if (wave == 0) {   // TENSORcnt is per-wave; wave0 issued the TDM ops
            if (step + 1 < nsteps) __builtin_amdgcn_s_wait_tensorcnt(1);
            else                   __builtin_amdgcn_s_wait_tensorcnt(0);
        }
#endif
        __syncthreads();   // current buffer visible to all waves

        __builtin_prefetch(arow0 + k0 + 64, 0, 1);
        __builtin_prefetch(arow1 + k0 + 64, 0, 1);
        bf16x16 a0 = load_a_frag(arow0 + k0, half);
        bf16x16 a1 = load_a_frag(arow1 + k0, half);
#pragma unroll
        for (int t = 0; t < 4; ++t) {
            bf16x16 bfrag = load_b_frag(&sB[cur][(t * 16 + l16) * 32], half);
            c[0][t] = wmma_bf16(a0, bfrag, c[0][t]);
            c[1][t] = wmma_bf16(a1, bfrag, c[1][t]);
        }
        __syncthreads();   // done reading sB[cur] before it is refilled
    }
#pragma unroll
    for (int ms = 0; ms < 2; ++ms)
#pragma unroll
        for (int t = 0; t < 4; ++t)
#pragma unroll
            for (int r = 0; r < 8; ++r)
                C[(size_t)(m0 + ms * 16 + r + 8 * half) * N + n0 + t * 16 + l16] =
                    c[ms][t][r];
}

// ---------------------------------------------------------------------------
// RMSNorm + RoPE. src = projection output (b,t,h,d) fp32, dst = (b,h,t,d) bf16
// ---------------------------------------------------------------------------
__global__ __launch_bounds__(256)
void rmsnorm_rope_kernel(const float* __restrict__ src, __bf16* __restrict__ dst,
                         const int* __restrict__ pos_ids, int nH) {
    const int lane = threadIdx.x & 31;
    const int wave = threadIdx.x >> 5;
    const int idx  = blockIdx.x * 8 + wave;      // (b*nH + h)*T + t
    const int t  = idx % T_SEQ;
    const int bh = idx / T_SEQ;
    const int h  = bh % nH;
    const int b  = bh / nH;

    const float* s = src + (size_t)(b * T_SEQ + t) * (nH * DH) + h * DH;
    float v[4];
    float ss = 0.0f;
#pragma unroll
    for (int e = 0; e < 4; ++e) { v[e] = s[lane + 32 * e]; ss += v[e] * v[e]; }
#pragma unroll
    for (int m = 1; m < 32; m <<= 1) ss += __shfl_xor(ss, m, 32);
    const float rms = rsqrtf(ss * (1.0f / 128.0f) + 1.1920929e-07f);

    const float p = (float)pos_ids[t];
    __bf16* d = dst + ((size_t)bh * T_SEQ + t) * DH;
#pragma unroll
    for (int pr = 0; pr < 2; ++pr) {
        const int j = lane + 32 * pr;                 // 0..63
        const float inv_freq = powf(10000.0f, -(float)j * (1.0f / 64.0f));
        float sn, cs;
        sincosf(p * inv_freq, &sn, &cs);
        const float x1 = v[pr] * rms;
        const float x2 = v[pr + 2] * rms;
        d[j]      = (__bf16)(x1 * cs - x2 * sn);
        d[j + 64] = (__bf16)(x2 * cs + x1 * sn);
    }
}

// ---------------------------------------------------------------------------
// V lerp with v1, store transposed: vt[(b,kv,d,t)] bf16
// ---------------------------------------------------------------------------
__global__ __launch_bounds__(256)
void v_merge_kernel(const float* __restrict__ Vf, const float* __restrict__ v1,
                    const float* __restrict__ lamb, __bf16* __restrict__ vt) {
    const int i = blockIdx.x * 256 + threadIdx.x;  // ((b*4+kv)*128 + d)*T + t
    const float L = lamb[0];
    const int t  = i % T_SEQ;
    const int r  = i / T_SEQ;
    const int d  = r % DH;
    const int bk = r / DH;               // b*4 + kv
    const int kv = bk % NKV;
    const int b  = bk / NKV;
    const float vp = Vf[(size_t)(b * T_SEQ + t) * (NKV * DH) + kv * DH + d];
    const float w1 = v1[((size_t)bk * T_SEQ + t) * DH + d];
    vt[i] = (__bf16)(vp + L * (w1 - vp));
}

// ---------------------------------------------------------------------------
// Gate: g[(b*16+h)*T + t] = sigmoid(sum_j x[b,t,j]*Wg[h,j]), j<16
// ---------------------------------------------------------------------------
__global__ __launch_bounds__(256)
void gate_kernel(const float* __restrict__ x, const float* __restrict__ Wg,
                 float* __restrict__ g) {
    const int i  = blockIdx.x * 256 + threadIdx.x;
    const int t  = i % T_SEQ;
    const int bh = i / T_SEQ;
    const int h  = bh % NH;
    const int b  = bh / NH;
    const float* xr = x + (size_t)(b * T_SEQ + t) * DM;
    const float* w  = Wg + h * NH;
    float s = 0.0f;
#pragma unroll
    for (int j = 0; j < NH; ++j) s += xr[j] * w[j];
    g[i] = 1.0f / (1.0f + expf(-s));
}

// ---------------------------------------------------------------------------
// Flash attention. One wave = one 16-row query tile. Block = 8 waves (one b,h).
// grid = (T/128, NH, B). s-loop in chunks of 32 keys; causal; online softmax.
// ---------------------------------------------------------------------------
__global__ __launch_bounds__(256)
void flash_attn_kernel(const __bf16* __restrict__ qb, const __bf16* __restrict__ kb,
                       const __bf16* __restrict__ vt, const float* __restrict__ g,
                       __bf16* __restrict__ yb) {
    __shared__ float ldsP[8][16 * 32];
    const int lane = threadIdx.x & 31;
    const int wave = threadIdx.x >> 5;
    const int half = lane >> 4;
    const int l16  = lane & 15;
    const int b  = blockIdx.z;
    const int h  = blockIdx.y;
    const int kv = h >> 2;                      // rep = NH/NKV = 4
    const int qt0 = (blockIdx.x * 8 + wave) * 16;
    const float scale = 0.08838834764831845f;   // 1/sqrt(128)

    const __bf16* qbase = qb + ((size_t)(b * NH + h) * T_SEQ) * DH;
    const __bf16* kbase = kb + ((size_t)(b * NKV + kv) * T_SEQ) * DH;
    const __bf16* vbase = vt + ((size_t)(b * NKV + kv) * DH) * T_SEQ;

    // Preload Q fragments: 4 d-chunks of 32
    bf16x16 qa[4];
    {
        const __bf16* row = qbase + (size_t)(qt0 + l16) * DH;
#pragma unroll
        for (int dc = 0; dc < 4; ++dc) qa[dc] = load_a_frag(row + dc * 32, half);
    }

    f32x8 acc[8];
#pragma unroll
    for (int dt = 0; dt < 8; ++dt)
#pragma unroll
        for (int r = 0; r < 8; ++r) acc[dt][r] = 0.0f;

    float mrun[8], lrun[8];
#pragma unroll
    for (int r = 0; r < 8; ++r) { mrun[r] = -1e30f; lrun[r] = 0.0f; }

    float* myP = &ldsP[wave][0];
    const int send = qt0 + 15;

    for (int s0 = 0; s0 <= send; s0 += 32) {
        // ---- S = Q @ K^T over 4 d-chunks; two 16-col score tiles ----
        f32x8 sc0, sc1;
#pragma unroll
        for (int r = 0; r < 8; ++r) { sc0[r] = 0.0f; sc1[r] = 0.0f; }
#pragma unroll
        for (int dc = 0; dc < 4; ++dc) {
            const __bf16* c0 = kbase + (size_t)(s0 + l16) * DH + dc * 32;
            const __bf16* c1 = kbase + (size_t)(s0 + 16 + l16) * DH + dc * 32;
            bf16x16 kf0 = load_b_frag(c0, half);
            bf16x16 kf1 = load_b_frag(c1, half);
            sc0 = wmma_bf16(qa[dc], kf0, sc0);
            sc1 = wmma_bf16(qa[dc], kf1, sc1);
        }
        // ---- scale, causal mask, online softmax ----
        float corr[8];
#pragma unroll
        for (int r = 0; r < 8; ++r) {
            const int row = qt0 + r + 8 * half;
            float v0 = sc0[r] * scale;
            float v1s = sc1[r] * scale;
            if (s0 + l16 > row)      v0 = -1e30f;
            if (s0 + 16 + l16 > row) v1s = -1e30f;
            float rm = fmaxf(v0, v1s);
            rm = fmaxf(rm, __shfl_xor(rm, 1, 32));
            rm = fmaxf(rm, __shfl_xor(rm, 2, 32));
            rm = fmaxf(rm, __shfl_xor(rm, 4, 32));
            rm = fmaxf(rm, __shfl_xor(rm, 8, 32));
            const float mnew = fmaxf(mrun[r], rm);
            corr[r] = __expf(mrun[r] - mnew);
            mrun[r] = mnew;
            const float p0 = __expf(v0 - mnew);
            const float p1 = __expf(v1s - mnew);
            myP[(r + 8 * half) * 32 + l16]      = p0;
            myP[(r + 8 * half) * 32 + 16 + l16] = p1;
            float rs = p0 + p1;
            rs += __shfl_xor(rs, 1, 32);
            rs += __shfl_xor(rs, 2, 32);
            rs += __shfl_xor(rs, 4, 32);
            rs += __shfl_xor(rs, 8, 32);
            lrun[r] = lrun[r] * corr[r] + rs;
        }
#pragma unroll
        for (int dt = 0; dt < 8; ++dt)
#pragma unroll
            for (int r = 0; r < 8; ++r) acc[dt][r] *= corr[r];

        // LDS store->load within wave: DS ops are in-order; wait for writeback
        asm volatile("s_wait_dscnt 0" ::: "memory");

        // ---- re-layout P (C layout -> A fragment) via LDS ----
        bf16x16 pa;
        {
            const float* prow = myP + l16 * 32;
#pragma unroll
            for (int i = 0; i < 16; ++i) {
                const int kk = (i < 8) ? (i + 8 * half) : (i + 8 + 8 * half);
                pa[i] = (__bf16)prow[kk];
            }
        }
        // ---- O += P @ V (V stored transposed (d,t): contiguous B frags) ----
#pragma unroll
        for (int dt = 0; dt < 8; ++dt) {
            const __bf16* vc = vbase + (size_t)(dt * 16 + l16) * T_SEQ + s0;
            bf16x16 vfrag = load_b_frag(vc, half);
            acc[dt] = wmma_bf16(pa, vfrag, acc[dt]);
        }
    }

    // ---- normalize, gate, store y (b, t, h*128 + d) bf16 ----
    float gfac[8];
#pragma unroll
    for (int r = 0; r < 8; ++r) {
        const int row = qt0 + r + 8 * half;
        gfac[r] = g[(size_t)(b * NH + h) * T_SEQ + row] / lrun[r];
    }
#pragma unroll
    for (int dt = 0; dt < 8; ++dt)
#pragma unroll
        for (int r = 0; r < 8; ++r) {
            const int row = qt0 + r + 8 * half;
            yb[(size_t)(b * T_SEQ + row) * DM + h * DH + dt * 16 + l16] =
                (__bf16)(acc[dt][r] * gfac[r]);
        }
}

// ---------------------------------------------------------------------------
// Launcher
// ---------------------------------------------------------------------------
extern "C" void kernel_launch(void* const* d_in, const int* in_sizes, int n_in,
                              void* d_out, int out_size, void* d_ws, size_t ws_size,
                              hipStream_t stream) {
    const float* x   = (const float*)d_in[0];
    const int*   pos = (const int*)  d_in[1];
    const float* v1  = (const float*)d_in[2];
    const float* Wq  = (const float*)d_in[3];
    const float* Wk  = (const float*)d_in[4];
    const float* Wv  = (const float*)d_in[5];
    const float* Wo  = (const float*)d_in[6];
    const float* Wg  = (const float*)d_in[7];
    const float* lam = (const float*)d_in[8];
    (void)in_sizes; (void)n_in; (void)out_size; (void)ws_size;

    char* ws = (char*)d_ws;
    size_t off = 0;
    auto take = [&](size_t bytes) -> char* {
        char* p = ws + off;
        off += (bytes + 255) & ~(size_t)255;
        return p;
    };
    __bf16* xb  = (__bf16*)take((size_t)4096 * 2048 * 2);   // x bf16 (M,K)
    __bf16* WqT = (__bf16*)take((size_t)2048 * 2048 * 2);   // Wq^T bf16 (N,K)
    __bf16* WkT = (__bf16*)take((size_t)512  * 2048 * 2);
    __bf16* WvT = (__bf16*)take((size_t)512  * 2048 * 2);
    __bf16* WoT = (__bf16*)take((size_t)2048 * 2048 * 2);
    float*  Qf  = (float*) take((size_t)4096 * 2048 * 4);   // Q proj fp32
    float*  Kf  = (float*) take((size_t)4096 * 512  * 4);
    float*  Vf  = (float*) take((size_t)4096 * 512  * 4);
    __bf16* kbq = (__bf16*)take((size_t)2 * NKV * T_SEQ * DH * 2); // k (b,kv,t,d)
    __bf16* vtb = (__bf16*)take((size_t)2 * NKV * T_SEQ * DH * 2); // v (b,kv,d,t)
    float*  gbuf= (float*) take((size_t)2 * NH * T_SEQ * 4);
    // Aliases with disjoint lifetimes:
    __bf16* qbq = (__bf16*)xb;   // xb dead after Q/K/V GEMMs; q (b,h,t,d)
    __bf16* yb  = (__bf16*)Qf;   // Qf dead after q rmsnorm/rope; y bf16 (M,K)

    // 1) precision conversion + weight transposes
    cvt_bf16_kernel<<<32768, 256, 0, stream>>>(x, xb, 4096 * 2048);
    transpose_bf16_kernel<<<16384, 256, 0, stream>>>(Wq, WqT, 2048, 2048);
    transpose_bf16_kernel<<<4096,  256, 0, stream>>>(Wk, WkT, 2048, 512);
    transpose_bf16_kernel<<<4096,  256, 0, stream>>>(Wv, WvT, 2048, 512);
    transpose_bf16_kernel<<<16384, 256, 0, stream>>>(Wo, WoT, 2048, 2048);

    // 2) QKV projections (WMMA bf16 GEMMs, TDM-staged B tiles)
    gemm_bf16_kernel<<<dim3(2048 / 64, 4096 / 256), 256, 0, stream>>>(xb, WqT, Qf, 4096, 2048, 2048);
    gemm_bf16_kernel<<<dim3(512  / 64, 4096 / 256), 256, 0, stream>>>(xb, WkT, Kf, 4096, 512,  2048);
    gemm_bf16_kernel<<<dim3(512  / 64, 4096 / 256), 256, 0, stream>>>(xb, WvT, Vf, 4096, 512,  2048);

    // 3) per-head rmsnorm + rope (q,k); v lerp + transpose; gates
    rmsnorm_rope_kernel<<<(2 * NH  * T_SEQ) / 8, 256, 0, stream>>>(Qf, qbq, pos, NH);
    rmsnorm_rope_kernel<<<(2 * NKV * T_SEQ) / 8, 256, 0, stream>>>(Kf, kbq, pos, NKV);
    v_merge_kernel<<<(2 * NKV * T_SEQ * DH) / 256, 256, 0, stream>>>(Vf, v1, lam, vtb);
    gate_kernel<<<(2 * NH * T_SEQ) / 256, 256, 0, stream>>>(x, Wg, gbuf);

    // 4) causal flash attention (WMMA)
    flash_attn_kernel<<<dim3(T_SEQ / 128, NH, 2), 256, 0, stream>>>(qbq, kbq, vtb, gbuf, yb);

    // 5) output projection -> d_out[0 : B*T*DM]
    gemm_bf16_kernel<<<dim3(2048 / 64, 4096 / 256), 256, 0, stream>>>(yb, WoT, (float*)d_out, 4096, 2048, 2048);

    // 6) second output of the tuple: v1 passthrough
    (void)hipMemcpyAsync((float*)d_out + (size_t)4096 * 2048, (void*)v1,
                         (size_t)2 * NKV * T_SEQ * DH * sizeof(float),
                         hipMemcpyDeviceToDevice, stream);
}
